// Student_LightGCL_73890617360945
// MI455X (gfx1250) — matrix operand order repaired and checked by reading, compiled, not run on hardware
//
#include <hip/hip_runtime.h>

typedef __attribute__((ext_vector_type(16))) _Float16 v16h;
typedef __attribute__((ext_vector_type(8)))  _Float16 v8h;
typedef __attribute__((ext_vector_type(8)))  float    v8f;
typedef __attribute__((ext_vector_type(2)))  int      v2i;

#define DDIM 64
#define ROWP 72   // LDS row pitch in halves (64 + 8 pad -> 36-dword stride, conflict-free)

// ---- CDNA5 async global->LDS staging (guarded; falls back to register staging) ----
#if defined(__has_builtin)
#if __has_builtin(__builtin_amdgcn_global_load_async_to_lds_b64) && \
    __has_builtin(__builtin_amdgcn_s_wait_asynccnt)
#define ASYNC_OK 1
#endif
#endif
#ifndef ASYNC_OK
#define ASYNC_OK 0
#endif

#define AS1 __attribute__((address_space(1)))
#define AS3 __attribute__((address_space(3)))

__device__ __forceinline__ void stage8(const _Float16* g, _Float16* l){
#if ASYNC_OK
  // generic->AS casts via integer: global generic addr == global addr,
  // LDS generic addr low 32 bits == LDS offset (ISA flat aperture rule)
  __builtin_amdgcn_global_load_async_to_lds_b64(
      (AS1 v2i*)(unsigned long long)g,
      (AS3 v2i*)(unsigned)(unsigned long long)l, 0, 0);
#else
  *(uint2*)l = *(const uint2*)g;
#endif
}

__device__ __forceinline__ void stage_commit(){
#if ASYNC_OK
  __builtin_amdgcn_s_wait_asynccnt(0);
#endif
  __syncthreads();
}

// ---------------- elementwise helpers ----------------

__global__ void k_zero(float* __restrict__ p, long n){
  long t = (long)blockIdx.x*blockDim.x + threadIdx.x;
  if (t < n) p[t] = 0.f;
}

__global__ void k_addcopy(float* __restrict__ out, float* __restrict__ cpy,
                          const float* __restrict__ a, const float* __restrict__ b, long n){
  long t = (long)blockIdx.x*blockDim.x + threadIdx.x;
  if (t < n){ float v = a[t]+b[t]; out[t]=v; cpy[t]=v; }
}

__global__ void k_add2(float* __restrict__ out, const float* __restrict__ a,
                       const float* __restrict__ b, long n){
  long t = (long)blockIdx.x*blockDim.x + threadIdx.x;
  if (t < n) out[t] = a[t]+b[t];
}

// ---------------- COO SpMM: Z[dst[e],:] += val[e]*E[src[e],:]; 16 threads/edge x 4 dims ----------------

__global__ void k_spmm(const float* __restrict__ vals, const int* __restrict__ dst,
                       const int* __restrict__ src, const float* __restrict__ E,
                       float* __restrict__ Z, int nnz){
  long t = (long)blockIdx.x*blockDim.x + threadIdx.x;
  int e = (int)(t >> 4);
  if (e >= nnz) return;
  int part = (int)(t & 15);
  float v = vals[e];
  size_t sb = (size_t)src[e]*DDIM + part*4;
  size_t db = (size_t)dst[e]*DDIM + part*4;
  float4 x = *(const float4*)(E + sb);
  unsafeAtomicAdd(&Z[db+0], v*x.x);   // hw global_atomic_add_f32 (L2 atomic units)
  unsafeAtomicAdd(&Z[db+1], v*x.y);
  unsafeAtomicAdd(&Z[db+2], v*x.z);
  unsafeAtomicAdd(&Z[db+3], v*x.w);
}

// ---------------- low-rank: P[r,d] = sum_i W[r,i]*E[i,d]  (320 blocks) ----------------

__global__ void k_lrproj(const float* __restrict__ W, const float* __restrict__ E,
                         float* __restrict__ P, int N){
  int r = blockIdx.x >> 6;
  int d = blockIdx.x & 63;
  const float* w = W + (size_t)r*N;
  float s = 0.f;
  for (int i = threadIdx.x; i < N; i += blockDim.x) s += w[i]*E[(size_t)i*DDIM + d];
  __shared__ float red[256];
  red[threadIdx.x] = s; __syncthreads();
  for (int st=128; st>0; st>>=1){
    if ((int)threadIdx.x < st) red[threadIdx.x] += red[threadIdx.x+st];
    __syncthreads();
  }
  if (threadIdx.x==0) P[blockIdx.x] = red[0];
}

// Gacc[u,d] += sum_r M5[u,r]*P[r,d]
__global__ void k_lrapply(const float* __restrict__ M5, const float* __restrict__ P,
                          float* __restrict__ Gacc, long n){
  long t = (long)blockIdx.x*blockDim.x + threadIdx.x;
  if (t >= n) return;
  long u = t >> 6; int d = (int)(t & 63);
  const float* m = M5 + u*5;
  float s = 0.f;
  #pragma unroll
  for (int r=0;r<5;++r) s += m[r]*P[r*DDIM + d];
  Gacc[t] += s;
}

// ---------------- combine: E=Eacc/3, G=Gacc/3, Eh=(f16)E, out = E + 0.02*(norm img + norm txt) ----------------

__global__ void k_combine(float* __restrict__ Eacc, float* __restrict__ Gacc,
                          const float* __restrict__ img, const float* __restrict__ txt,
                          _Float16* __restrict__ Eh, float* __restrict__ outp, int rows){
  int lane = threadIdx.x & 31;
  int wid  = threadIdx.x >> 5;
  int row  = blockIdx.x*8 + wid;
  if (row >= rows) return;
  size_t b = (size_t)row*DDIM;
  int d0 = lane, d1 = lane+32;
  float i0=img[b+d0], i1=img[b+d1], t0=txt[b+d0], t1=txt[b+d1];
  float si = i0*i0 + i1*i1;
  float st = t0*t0 + t1*t1;
  #pragma unroll
  for (int off=16; off>0; off>>=1){
    si += __shfl_xor(si, off, 32);
    st += __shfl_xor(st, off, 32);
  }
  float ni = 1.f/fmaxf(sqrtf(si), 1e-12f);
  float nt = 1.f/fmaxf(sqrtf(st), 1e-12f);
  const float c = 0.02f, third = (1.f/3.f);
  float e0 = Eacc[b+d0]*third, e1 = Eacc[b+d1]*third;
  float g0 = Gacc[b+d0]*third, g1 = Gacc[b+d1]*third;
  Eacc[b+d0]=e0; Eacc[b+d1]=e1;
  Gacc[b+d0]=g0; Gacc[b+d1]=g1;
  Eh[b+d0]=(_Float16)e0; Eh[b+d1]=(_Float16)e1;
  outp[b+d0] = e0 + c*(i0*ni) + c*(t0*nt);
  outp[b+d1] = e1 + c*(i1*ni) + c*(t1*nt);
}

// ---------------- pos term ----------------

__global__ void k_pos(const int* __restrict__ ids, int cnt, const float* __restrict__ G,
                      const float* __restrict__ E, float* acc){
  __shared__ float red[256];
  int t = blockIdx.x*256 + threadIdx.x;
  float v = 0.f;
  if (t < cnt){
    size_t b = (size_t)ids[t]*DDIM;
    float s = 0.f;
    #pragma unroll
    for (int d=0; d<DDIM; ++d) s += G[b+d]*E[b+d];
    s *= 5.0f; // 1/TEMP
    v = fminf(fmaxf(s, -5.f), 5.f);
  }
  red[threadIdx.x]=v; __syncthreads();
  for (int st=128; st>0; st>>=1){
    if ((int)threadIdx.x < st) red[threadIdx.x] += red[threadIdx.x+st];
    __syncthreads();
  }
  if (threadIdx.x==0) atomicAdd(acc, red[0]);
}

__global__ void k_gather_h(const int* __restrict__ ids, int cnt,
                           const float* __restrict__ G, _Float16* __restrict__ A){
  long t = (long)blockIdx.x*blockDim.x + threadIdx.x;
  if (t >= (long)cnt*DDIM) return;
  long rr = t >> 6; int d = (int)(t & 63);
  A[t] = (_Float16)G[(size_t)ids[rr]*DDIM + d];
}

// ---------------- WMMA score GEMM + row sum-exp partials ----------------
// Block = 8 waves; each wave owns one 16-row tile; all 8 share the 16x64 f16 B tile
// staged in LDS (async copy, double buffered). K=64 -> two v_wmma_f32_16x16x32_f16.
// A frag: lane m=lane%16, halves j -> K=(j/8)*16+(lane/16)*8+(j%8)
// B frag: lane n=lane%16, halves j -> K=(j/8)*16+(lane/16)*8+(j%8)  (B[k][n]=E[n][k])
// D frag: lane n=lane%16, vgpr r -> m=r+8*(lane/16)
// Scores are bounded (|dot|/T <~ 15) so we accumulate sum(exp(x)) directly (no rescale).

__global__ void k_wmma_lse(const _Float16* __restrict__ A, const _Float16* __restrict__ Bm,
                           int N, int Mrows, int chunkCols, float* __restrict__ psum){
  __shared__ __align__(16) _Float16 sB[2][16*ROWP];

  const int tid  = threadIdx.x;
  const int wid  = tid >> 5;
  const int lane = tid & 31;
  const int hi   = lane >> 4;
  const int lo16 = lane & 15;
  const int Mtiles = Mrows >> 4;
  const int tileRaw = blockIdx.x*8 + wid;
  const bool active = tileRaw < Mtiles;
  const int tile = active ? tileRaw : (Mtiles-1);

  // per-wave A fragments (registers, reused across the whole column sweep)
  const _Float16* Ar = A + (size_t)(tile*16 + lo16)*DDIM;
  v8h al0 = *(const v8h*)(Ar +  0 + hi*8);
  v8h ah0 = *(const v8h*)(Ar + 16 + hi*8);
  v8h al1 = *(const v8h*)(Ar + 32 + hi*8);
  v8h ah1 = *(const v8h*)(Ar + 48 + hi*8);
  v16h a0 = __builtin_shufflevector(al0, ah0, 0,1,2,3,4,5,6,7,8,9,10,11,12,13,14,15);
  v16h a1 = __builtin_shufflevector(al1, ah1, 0,1,2,3,4,5,6,7,8,9,10,11,12,13,14,15);

  const int n0 = blockIdx.y*chunkCols;
  int n1 = n0 + chunkCols; if (n1 > N) n1 = N;   // N multiple of 16
  const int ntiles = (n1 - n0) >> 4;

  // cooperative staging: 256 threads x 8B = 16 rows x 128B tile
  const int srow = tid >> 4;        // 0..15
  const int sseg = (tid & 15) * 4;  // half offset, 8B units

  stage8(Bm + (size_t)(n0 + srow)*DDIM + sseg, &sB[0][srow*ROWP + sseg]);
  stage_commit();

  float sm[8];
  #pragma unroll
  for (int r=0;r<8;++r) sm[r] = 0.f;
  const float invT = 5.0f;

  for (int i=0; i<ntiles; ++i){
    const int cur = i & 1;
    if (i+1 < ntiles)
      stage8(Bm + (size_t)(n0 + (i+1)*16 + srow)*DDIM + sseg, &sB[cur^1][srow*ROWP + sseg]);

    const _Float16* br = &sB[cur][lo16*ROWP + hi*8];
    v8h q0 = *(const v8h*)(br);        // K hi*8 .. +7
    v8h q1 = *(const v8h*)(br + 16);   // K 16+hi*8 .. +7
    v8h q2 = *(const v8h*)(br + 32);   // K 32+hi*8 ..
    v8h q3 = *(const v8h*)(br + 48);
    v16h b0 = __builtin_shufflevector(q0, q1, 0,1,2,3,4,5,6,7,8,9,10,11,12,13,14,15);
    v16h b1 = __builtin_shufflevector(q2, q3, 0,1,2,3,4,5,6,7,8,9,10,11,12,13,14,15);

    v8f c = {};
    c = __builtin_amdgcn_wmma_f32_16x16x32_f16(false, a0, false, b0, (short)0, c, false, false);
    c = __builtin_amdgcn_wmma_f32_16x16x32_f16(false, a1, false, b1, (short)0, c, false, false);
    #pragma unroll
    for (int r=0;r<8;++r) sm[r] += __expf(c[r]*invT);

    if (i+1 < ntiles) stage_commit();   // next buffer ready + all reads of cur done
  }

  // sum across the 16 lanes of each half (same rows, distinct columns)
  #pragma unroll
  for (int r=0;r<8;++r){
    #pragma unroll
    for (int off=1; off<16; off<<=1) sm[r] += __shfl_xor(sm[r], off, 32);
  }
  if (active && lo16 == 0){
    #pragma unroll
    for (int r=0;r<8;++r){
      int row = tile*16 + hi*8 + r;
      psum[(size_t)blockIdx.y*Mrows + row] = sm[r];
    }
  }
}

// merge per-chunk partial sums -> logsumexp per row -> sum into acc
__global__ void k_merge(const float* __restrict__ psum, int Mrows, int nch, float* acc){
  __shared__ float red[256];
  int t = blockIdx.x*256 + threadIdx.x;
  float lse = 0.f;
  if (t < Mrows){
    float S = 0.f;
    for (int c=0;c<nch;++c) S += psum[(size_t)c*Mrows + t];
    lse = __logf(S);
  }
  red[threadIdx.x]=lse; __syncthreads();
  for (int st=128; st>0; st>>=1){
    if ((int)threadIdx.x < st) red[threadIdx.x] += red[threadIdx.x+st];
    __syncthreads();
  }
  if (threadIdx.x==0) atomicAdd(acc, red[0]);
}

__global__ void k_final(const float* __restrict__ acc, float* __restrict__ outloss,
                        float cu, float ci){
  if (threadIdx.x==0){
    float posSum = acc[0]/cu + acc[1]/ci;
    float negSum = acc[2]/cu + acc[3]/ci;
    outloss[0] = (negSum - posSum) * 4.0f;   // (len(G_u_list)+1) == 4
  }
}

// ---------------- host ----------------

extern "C" void kernel_launch(void* const* d_in, const int* in_sizes, int n_in,
                              void* d_out, int out_size, void* d_ws, size_t ws_size,
                              hipStream_t stream){
  const float* user_emb      = (const float*)d_in[0];
  const float* item_emb      = (const float*)d_in[1];
  const float* user_emb_pre  = (const float*)d_in[2];
  const float* item_emb_pre  = (const float*)d_in[3];
  const float* image_item    = (const float*)d_in[4];
  const float* text_item     = (const float*)d_in[5];
  const float* image_user    = (const float*)d_in[6];
  const float* text_user     = (const float*)d_in[7];
  const int*   adj_rows      = (const int*)d_in[8];
  const int*   adj_cols      = (const int*)d_in[9];
  const float* adj_vals      = (const float*)d_in[10];
  const float* ut            = (const float*)d_in[11];
  const float* vt            = (const float*)d_in[12];
  const float* u_mul_s       = (const float*)d_in[13];
  const float* v_mul_s       = (const float*)d_in[14];
  const int*   uids          = (const int*)d_in[15];
  const int*   iids          = (const int*)d_in[16];

  const int NU  = in_sizes[0]/DDIM;
  const int NI  = in_sizes[1]/DDIM;
  const int NNZ = in_sizes[8];
  const int CU  = in_sizes[15];
  const int CI  = in_sizes[16];
  const long nu = (long)NU*DDIM, ni = (long)NI*DDIM;

  float* ws = (float*)d_ws;
  size_t off = 0;
  auto alloc = [&](size_t n){ float* p = ws + off; off += n; return p; };
  float* Eu0   = alloc(nu);
  float* Ei0   = alloc(ni);
  float* Zu1   = alloc(nu);
  float* Zi1   = alloc(ni);          // Zu1,Zi1 contiguous (zeroed together)
  float* EaccU = alloc(nu);
  float* EaccI = alloc(ni);
  float* GaccU = alloc(nu);
  float* GaccI = alloc(ni);
  float* vtE   = alloc(5*DDIM);
  float* utE   = alloc(5*DDIM);
  float* acc   = alloc(8);
  _Float16* EuH = (_Float16*)alloc(nu/2);
  _Float16* EiH = (_Float16*)alloc(ni/2);
  _Float16* AuH = (_Float16*)alloc((size_t)CU*DDIM/2);
  _Float16* AiH = (_Float16*)alloc((size_t)CI*DDIM/2);
  const int CHUNK = 2048;
  const int nchU = (NU + CHUNK-1)/CHUNK;
  const int nchI = (NI + CHUNK-1)/CHUNK;
  float* psumU = alloc((size_t)nchU*CU);
  float* psumI = alloc((size_t)nchI*CI);

  float* out_user = (float*)d_out;
  float* out_item = out_user + nu;
  float* out_loss = out_item + ni;

  const int B = 256;
  const int gu = (int)((nu + B-1)/B), gi = (int)((ni + B-1)/B);

  // init / clear accumulation state (must run every call; ws is not re-poisoned)
  k_zero<<<(int)((nu+ni+B-1)/B), B, 0, stream>>>(Zu1, nu+ni);
  k_zero<<<1, 32, 0, stream>>>(acc, 8);
  k_addcopy<<<gu, B, 0, stream>>>(Eu0, GaccU, user_emb_pre, user_emb, nu);
  k_addcopy<<<gi, B, 0, stream>>>(Ei0, GaccI, item_emb_pre, item_emb, ni);

  // ---- layer 1 ----
  int spgrid = (int)(((long)NNZ*16 + B-1)/B);
  k_spmm<<<spgrid, B, 0, stream>>>(adj_vals, adj_rows, adj_cols, Ei0, Zu1, NNZ);
  k_spmm<<<spgrid, B, 0, stream>>>(adj_vals, adj_cols, adj_rows, Eu0, Zi1, NNZ);
  k_lrproj<<<5*DDIM, B, 0, stream>>>(vt, Ei0, vtE, NI);
  k_lrproj<<<5*DDIM, B, 0, stream>>>(ut, Eu0, utE, NU);
  k_lrapply<<<gu, B, 0, stream>>>(u_mul_s, vtE, GaccU, nu);
  k_lrapply<<<gi, B, 0, stream>>>(v_mul_s, utE, GaccI, ni);

  // Eacc = Eu0 + Zu1 ; layer-2 SpMM accumulates Zu2 straight into Eacc
  k_add2<<<gu, B, 0, stream>>>(EaccU, Eu0, Zu1, nu);
  k_add2<<<gi, B, 0, stream>>>(EaccI, Ei0, Zi1, ni);

  // ---- layer 2 ----
  k_spmm<<<spgrid, B, 0, stream>>>(adj_vals, adj_rows, adj_cols, Zi1, EaccU, NNZ);
  k_spmm<<<spgrid, B, 0, stream>>>(adj_vals, adj_cols, adj_rows, Zu1, EaccI, NNZ);
  k_lrproj<<<5*DDIM, B, 0, stream>>>(vt, Zi1, vtE, NI);
  k_lrproj<<<5*DDIM, B, 0, stream>>>(ut, Zu1, utE, NU);
  k_lrapply<<<gu, B, 0, stream>>>(u_mul_s, vtE, GaccU, nu);
  k_lrapply<<<gi, B, 0, stream>>>(v_mul_s, utE, GaccI, ni);

  // ---- combine: E=/3, G=/3, f16 staging, user_out/item_out ----
  k_combine<<<(NU+7)/8, B, 0, stream>>>(EaccU, GaccU, image_user, text_user, EuH, out_user, NU);
  k_combine<<<(NI+7)/8, B, 0, stream>>>(EaccI, GaccI, image_item, text_item, EiH, out_item, NI);

  // ---- loss ----
  k_pos<<<(CU+B-1)/B, B, 0, stream>>>(uids, CU, GaccU, EaccU, acc+0);
  k_pos<<<(CI+B-1)/B, B, 0, stream>>>(iids, CI, GaccI, EaccI, acc+1);

  k_gather_h<<<(int)(((long)CU*DDIM + B-1)/B), B, 0, stream>>>(uids, CU, GaccU, AuH);
  k_gather_h<<<(int)(((long)CI*DDIM + B-1)/B), B, 0, stream>>>(iids, CI, GaccI, AiH);

  dim3 gWU((CU/16 + 7)/8, nchU), gWI((CI/16 + 7)/8, nchI);
  k_wmma_lse<<<gWU, B, 0, stream>>>(AuH, EuH, NU, CU, CHUNK, psumU);
  k_wmma_lse<<<gWI, B, 0, stream>>>(AiH, EiH, NI, CI, CHUNK, psumI);

  k_merge<<<(CU+B-1)/B, B, 0, stream>>>(psumU, CU, nchU, acc+2);
  k_merge<<<(CI+B-1)/B, B, 0, stream>>>(psumI, CI, nchI, acc+3);

  k_final<<<1, 1, 0, stream>>>(acc, out_loss, (float)CU, (float)CI);
}
